// DiffusionStack_3685081940008
// MI455X (gfx1250) — compile-verified, matrix operand order checked by reading, not run on hardware
//
#include <hip/hip_runtime.h>
#include <hip/hip_bf16.h>

#define N_     1024
#define D_     384
#define K_     64
#define A_     5
#define BINS_  64
#define DEPTH_ 4
#define C3_    15   // A*3
#define NB_    64   // output columns per block (4 WMMA n-tiles)
#define PSTR_  68   // padded row stride (u32 words) for packed-B LDS (bank-conflict free)

typedef __attribute__((ext_vector_type(16))) __bf16       v16bf;
typedef __attribute__((ext_vector_type(8)))  float        v8f;
typedef __attribute__((ext_vector_type(8)))  unsigned int v8u;
typedef unsigned int u32x4 __attribute__((ext_vector_type(4)));
typedef int          i32x4 __attribute__((ext_vector_type(4)));
typedef int          i32x8 __attribute__((ext_vector_type(8)));

#if __has_builtin(__builtin_amdgcn_tensor_load_to_lds) && \
    __has_builtin(__builtin_amdgcn_s_wait_tensorcnt)
#define HAVE_TDM 1
#endif

// ---------- helpers ----------
__device__ __forceinline__ unsigned int hash_u32(unsigned int x) {
  x ^= x >> 16; x *= 0x7feb352dU;
  x ^= x >> 15; x *= 0x846ca68bU;
  x ^= x >> 16; return x;
}

__device__ __forceinline__ float gumbel_noise(int t, int i, int j) {
  unsigned int h = hash_u32((unsigned int)(i * N_ + j) ^ (0x9E3779B9u * (unsigned int)(t + 1)));
  float u = (float)(h >> 8) * (1.0f / 16777216.0f);        // [0,1)
  return -__logf(-__logf(u + 1e-6f) + 1e-6f);
}

// gelu (tanh approx) in sigmoid form: branch-free v_exp + v_rcp
__device__ __forceinline__ float gelu_f(float x) {
  float z = 1.5957691216057308f * (x + 0.044715f * x * x * x);
  return x * __frcp_rn(1.0f + __expf(-z));
}

// ---------- Kernel A: iteration-invariant base distance matrix ----------
__global__ void __launch_bounds__(256)
k_base(const float* __restrict__ distogram, const float* __restrict__ prev_pos,
       const int* __restrict__ resi, const int* __restrict__ chain,
       const int* __restrict__ batch, float* __restrict__ base)
{
  const int pair = blockIdx.x * 8 + (threadIdx.x >> 5);
  const int lane = threadIdx.x & 31;
  const int i = pair >> 10, j = pair & (N_ - 1);
  const float INF = __builtin_inff();

  const float* row = distogram + (size_t)pair * BINS_;
  __builtin_prefetch(distogram + (size_t)(pair + 8) * BINS_, 0, 1);
  float x0 = row[lane], x1 = row[lane + 32];

  float m = fmaxf(x0, x1);
  for (int off = 16; off; off >>= 1) m = fmaxf(m, __shfl_xor(m, off, 32));

  const float step = 22.0f / (float)BINS_;
  float c0 = step * (float)lane        + 0.5f * step;
  float c1 = step * (float)(lane + 32) + 0.5f * step;
  float e0 = __expf(x0 - m), e1 = __expf(x1 - m);
  float s = e0 + e1, w = e0 * c0 + e1 * c1;
  for (int off = 16; off; off >>= 1) { s += __shfl_xor(s, off, 32); w += __shfl_xor(w, off, 32); }

  if (lane == 0) {
    float md = w / s;
    float d  = (md < 8.0f) ? md : INF;
    bool sb = (batch[i] == batch[j]);
    if (sb && (chain[i] == chain[j]))
      d = fminf(d, fabsf((float)(resi[i] - resi[j])) * 3.81f);
    float dx = prev_pos[i * C3_ + 3] - prev_pos[j * C3_ + 3];
    float dy = prev_pos[i * C3_ + 4] - prev_pos[j * C3_ + 4];
    float dz = prev_pos[i * C3_ + 5] - prev_pos[j * C3_ + 5];
    d = fminf(d, sqrtf(dx * dx + dy * dy + dz * dz));
    if (!sb) d = INF;
    base[pair] = d;
  }
}

// ---------- Kernel B: per-row randomized distance + top-K ----------
__global__ void __launch_bounds__(256)
k_topk(const float* __restrict__ base, const float* __restrict__ pos,
       const int* __restrict__ batch, const unsigned char* __restrict__ mask,
       int t, int* __restrict__ idx, float* __restrict__ cnt)
{
  __shared__ float rd[N_];
  __shared__ unsigned long long red[256];
  const int i   = blockIdx.x;
  const int tid = threadIdx.x;
  const float INF = __builtin_inff();

  const int  bi = batch[i];
  const bool mi = mask[i] != 0;
  const float cax = pos[i * C3_ + 3], cay = pos[i * C3_ + 4], caz = pos[i * C3_ + 5];

  for (int j = tid; j < N_; j += 256) {
    float d  = base[(size_t)i * N_ + j];
    float dx = cax - pos[j * C3_ + 3];
    float dy = cay - pos[j * C3_ + 4];
    float dz = caz - pos[j * C3_ + 5];
    d = fminf(d, sqrtf(dx * dx + dy * dy + dz * dz));
    bool sb = (batch[j] == bi);
    float v = INF;
    if (sb && mi && mask[j]) v = 3.0f * d - gumbel_noise(t, i, j);  // -(log_p + g)
    rd[j] = v;
  }
  __syncthreads();

  int count = 0;
  for (int k = 0; k < K_; ++k) {
    float best = INF; int bj = 0;
    for (int j = tid; j < N_; j += 256) {
      float v = rd[j];
      if (v < best) { best = v; bj = j; }
    }
    unsigned int fb  = __float_as_uint(best);
    unsigned int key = (fb & 0x80000000u) ? ~fb : (fb | 0x80000000u);
    red[tid] = ((unsigned long long)key << 32) | (unsigned int)bj;
    __syncthreads();
    for (int s2 = 128; s2; s2 >>= 1) {
      if (tid < s2) {
        unsigned long long o = red[tid + s2];
        if (o < red[tid]) red[tid] = o;
      }
      __syncthreads();
    }
    unsigned long long top = red[0];
    __syncthreads();
    unsigned int tkey = (unsigned int)(top >> 32);
    int  jwin = (int)(top & 0xFFFFFFFFu);
    unsigned int fbw = (tkey & 0x80000000u) ? (tkey & 0x7FFFFFFFu) : ~tkey;
    bool fin = __uint_as_float(fbw) < INF;
    if (tid == 0) {
      idx[i * K_ + k] = fin ? jwin : -1;
      rd[jwin] = INF;
    }
    if (fin) ++count;
    __syncthreads();
  }
  if (tid == 0) cnt[i] = fmaxf((float)count, 1.0f);
}

// ---------- Kernel C: neighbor gather + mean ----------
__global__ void __launch_bounds__(128)
k_gather(const float* __restrict__ local, const int* __restrict__ idx,
         const float* __restrict__ cnt, float* __restrict__ msg)
{
  __shared__ int sidx[K_];
  const int i = blockIdx.x;
  if (threadIdx.x < K_) sidx[threadIdx.x] = idx[i * K_ + threadIdx.x];
  __syncthreads();
  const float inv = 1.0f / cnt[i];
  for (int d = threadIdx.x; d < D_; d += 128) {
    float acc = 0.0f;
    for (int k = 0; k < K_; ++k) {
      int j = sidx[k];
      if (j >= 0) acc += local[(size_t)j * D_ + d];
    }
    msg[(size_t)i * D_ + d] = acc * inv;
  }
}

// ---------- TDM: stage Wn[:, n0:n0+64] (384x64 fp32 tile) to LDS offset 0 ----------
#ifdef HAVE_TDM
__device__ __forceinline__ void tdm_load_wn_strip(const float* gsrc) {
  unsigned long long ga = (unsigned long long)(size_t)gsrc;
  u32x4 g0;
  g0[0] = 1u;                                   // count=1, user mode
  g0[1] = 0u;                                   // lds_addr = 0 (Bs32 at LDS offset 0)
  g0[2] = (unsigned)(ga & 0xFFFFFFFFu);         // global_addr[31:0]
  g0[3] = (unsigned)(ga >> 32) | (2u << 30);    // global_addr[56:32] | type=2
  i32x8 g1;
  g1[0] = (int)(2u << 16);                      // workgroup_mask=0, data_size=2 (4B)
  g1[1] = (int)(384u << 16);                    // tensor_dim0 (low 16b at bits 63:48)
  g1[2] = (int)(384u << 16);                    // tensor_dim1 (low 16b at bits 95:80)
  g1[3] = (int)((unsigned)NB_ << 16);           // tile_dim0 = 64 (bits 127:112)
  g1[4] = 384;                                  // tile_dim1 = 384 (bits 143:128)
  g1[5] = 384;                                  // tensor_dim0_stride low32
  g1[6] = 0;
  g1[7] = 0;
  i32x4 z4 = {0, 0, 0, 0};
#if defined(__clang_major__) && __clang_major__ >= 23
  i32x8 z8 = {0, 0, 0, 0, 0, 0, 0, 0};
  __builtin_amdgcn_tensor_load_to_lds(g0, g1, z4, z4, z8, 0);
#else
  __builtin_amdgcn_tensor_load_to_lds(g0, g1, z4, z4, 0);
#endif
  __builtin_amdgcn_s_wait_tensorcnt(0);
}
#endif

// ---------- Kernel D: local += gelu(msg @ Wn), bf16 WMMA, 4-way A reuse ----------
// Block = 8 waves x 16 rows (128 rows) x 64 columns (4 n-tiles per wave).
// Wn strip TDM-staged fp32 -> repacked once to bf16 K-pairs in LDS -> inner loop is
// pure ds_load_b32 + v_wmma (48 WMMAs/wave, A fragments loaded once per kk).
__global__ void __launch_bounds__(256)
k_gemm_gelu(const float* __restrict__ msg, const float* __restrict__ Wn,
            float* __restrict__ local)
{
  __shared__ float        Bs32[D_ * NB_];          // 96 KB fp32 strip (TDM dest, offset 0)
  __shared__ unsigned int Bp[(D_ / 2) * PSTR_];    // 52 KB packed bf16 K-pairs
  const int wave = threadIdx.x >> 5;
  const int lane = threadIdx.x & 31;
  const int n0   = blockIdx.x * NB_;
  const int m0   = blockIdx.y * 128 + wave * 16;
  const int half = lane >> 4;                      // 0: lanes 0-15, 1: lanes 16-31
  const int lm   = lane & 15;

  // Never-executed, statically opaque stores: the TDM intrinsic's LDS writes are
  // not modeled against these globals, and without any visible store LLVM folds
  // every LDS load to undef (observed: B operand became garbage and the 4 acc
  // chains CSE'd to one). A store instruction in IR blocks that transform.
  if (msg == (const float*)0) Bs32[0] = -1.0f;
  if (msg == (const float*)4) Bp[0]   = 1u;

#ifdef HAVE_TDM
  if (threadIdx.x < 32) tdm_load_wn_strip(Wn + n0);   // wave 0 issues + waits TENSORcnt
  __syncthreads();
  asm volatile("" ::: "memory");                   // TDM wrote LDS
#else
  for (int e = threadIdx.x; e < D_ * NB_; e += 256) {
    int k = e >> 6, n = e & (NB_ - 1);
    Bs32[e] = Wn[(size_t)k * D_ + n0 + n];
  }
  __syncthreads();
#endif

  // repack: Bp[p*PSTR_+n] = {bf16(Wn[2p][n]), bf16(Wn[2p+1][n])}  (one word = one K-pair)
  for (int e = threadIdx.x; e < (D_ / 2) * NB_; e += 256) {
    int pIdx = e >> 6, n = e & (NB_ - 1);
    union { __bf16 b[2]; unsigned int u; } pk;
    pk.b[0] = (__bf16)Bs32[(2 * pIdx)     * NB_ + n];
    pk.b[1] = (__bf16)Bs32[(2 * pIdx + 1) * NB_ + n];
    Bp[pIdx * PSTR_ + n] = pk.u;
  }
  __syncthreads();

  const float* arow = msg + (size_t)(m0 + lm) * D_;

  v8f acc[4] = {v8f{}, v8f{}, v8f{}, v8f{}};
#pragma unroll
  for (int kk = 0; kk < D_; kk += 32) {
    v16bf a;
#pragma unroll
    for (int h = 0; h < 8; ++h) {
      // 16-bit A 16x32 layout: VGPR h holds K pair; lanes>=16 +8; VGPRs 4-7 +16 (kb even)
      int kb = kk + ((h < 4) ? 2 * h : 16 + 2 * (h - 4)) + half * 8;
      float2 av = *(const float2*)(arow + kb);
      a[2 * h] = (__bf16)av.x; a[2 * h + 1] = (__bf16)av.y;
    }
#pragma unroll
    for (int nt = 0; nt < 4; ++nt) {
      v8u bu;
#pragma unroll
      for (int h = 0; h < 8; ++h) {
        int kb = kk + ((h < 4) ? 2 * h : 16 + 2 * (h - 4)) + half * 8;
        bu[h] = Bp[(kb >> 1) * PSTR_ + nt * 16 + lm];   // packed pair {kb, kb+1}
      }
      v16bf b = __builtin_bit_cast(v16bf, bu);
      acc[nt] = __builtin_amdgcn_wmma_f32_16x16x32_bf16(false, a, false, b,
                                                        (short)0, acc[nt], false, false);
    }
  }

#pragma unroll
  for (int nt = 0; nt < 4; ++nt) {
#pragma unroll
    for (int r = 0; r < 8; ++r) {
      int m = m0 + r + half * 8;                  // C/D: lanes 16-31 start at M=8
      int n = n0 + nt * 16 + lm;
      local[(size_t)m * D_ + n] += gelu_f(acc[nt][r]);
    }
  }
}

// ---------- Kernel E: pos += 0.1*(local @ Wp)*mask ; record trajectory ----------
__global__ void __launch_bounds__(128)
k_pos(const float* __restrict__ local, const float* __restrict__ Wp,
      const unsigned char* __restrict__ mask, float* __restrict__ pos,
      float* __restrict__ traj, int t)
{
  __shared__ float row[D_];
  const int i = blockIdx.x;
  for (int d = threadIdx.x; d < D_; d += 128) row[d] = local[(size_t)i * D_ + d];
  __syncthreads();
  if (threadIdx.x < C3_) {
    int c = threadIdx.x;
    float acc = 0.0f;
    for (int d = 0; d < D_; ++d) acc += row[d] * Wp[d * C3_ + c];
    float mf = mask[i] ? 1.0f : 0.0f;
    float p  = pos[i * C3_ + c] + 0.1f * acc * mf;
    pos[i * C3_ + c] = p;
    traj[(size_t)t * N_ * C3_ + i * C3_ + c] = p;
  }
}

// ---------- launch ----------
extern "C" void kernel_launch(void* const* d_in, const int* in_sizes, int n_in,
                              void* d_out, int out_size, void* d_ws, size_t ws_size,
                              hipStream_t stream) {
  (void)in_sizes; (void)n_in; (void)out_size; (void)ws_size;
  const float* local_in  = (const float*)d_in[0];
  const float* pos_in    = (const float*)d_in[1];
  const float* prev_pos  = (const float*)d_in[2];
  const float* distogram = (const float*)d_in[3];
  const float* Wn        = (const float*)d_in[4];
  const float* Wp        = (const float*)d_in[5];
  const int*   resi      = (const int*)d_in[6];
  const int*   chain     = (const int*)d_in[7];
  const int*   batch     = (const int*)d_in[8];
  const unsigned char* mask = (const unsigned char*)d_in[9];

  char* p = (char*)d_ws;
  float* base      = (float*)p; p += (size_t)N_ * N_ * sizeof(float);
  float* local_cur = (float*)p; p += (size_t)N_ * D_ * sizeof(float);
  float* msg       = (float*)p; p += (size_t)N_ * D_ * sizeof(float);
  float* pos_cur   = (float*)p; p += (size_t)N_ * C3_ * sizeof(float);
  int*   idx       = (int*)p;   p += (size_t)N_ * K_ * sizeof(int);
  float* cnt       = (float*)p; p += (size_t)N_ * sizeof(float);

  float* out_local = (float*)d_out;
  float* out_pos   = out_local + (size_t)N_ * D_;
  float* traj      = out_pos + (size_t)N_ * C3_;

  hipMemcpyAsync(local_cur, local_in, (size_t)N_ * D_ * sizeof(float),
                 hipMemcpyDeviceToDevice, stream);
  hipMemcpyAsync(pos_cur, pos_in, (size_t)N_ * C3_ * sizeof(float),
                 hipMemcpyDeviceToDevice, stream);

  // Iteration-invariant heavy term: one pass over the 268 MB distogram.
  k_base<<<(N_ * N_) / 8, 256, 0, stream>>>(distogram, prev_pos, resi, chain, batch, base);

  for (int t = 0; t < DEPTH_; ++t) {
    k_topk<<<N_, 256, 0, stream>>>(base, pos_cur, batch, mask, t, idx, cnt);
    k_gather<<<N_, 128, 0, stream>>>(local_cur, idx, cnt, msg);
    k_gemm_gelu<<<dim3(D_ / NB_, N_ / 128), 256, 0, stream>>>(msg, Wn, local_cur);
    k_pos<<<N_, 128, 0, stream>>>(local_cur, Wp, mask, pos_cur, traj, t);
  }

  hipMemcpyAsync(out_local, local_cur, (size_t)N_ * D_ * sizeof(float),
                 hipMemcpyDeviceToDevice, stream);
  hipMemcpyAsync(out_pos, pos_cur, (size_t)N_ * C3_ * sizeof(float),
                 hipMemcpyDeviceToDevice, stream);
}
